// PerformerAttentionCore_84524956385939
// MI455X (gfx1250) — matrix operand order repaired and checked by reading, not compiled, not run on hardware
//
#include <hip/hip_runtime.h>
#include <stdint.h>

// ---------------- problem constants ----------------
#define Bq 2
#define Hh 8
#define Nn 2048
#define Dd 32
#define Mm 64
#define CC 64               // chunk length
#define NC (Nn / CC)        // 32 chunks per head
#define BH (Bq * Hh)        // 16 head-slots

typedef __bf16 bf16;
typedef __attribute__((ext_vector_type(16))) __bf16 v16bf;
typedef __attribute__((ext_vector_type(8)))  __bf16 v8bf;
typedef __attribute__((ext_vector_type(8)))  float  v8f;

// ---- CDNA5 async global->LDS staging (ASYNCcnt-tracked DMA path) ----
__device__ __forceinline__ void async_b128(uint32_t lds_byte_off, const void* gptr) {
  asm volatile("global_load_async_to_lds_b128 %0, %1, off"
               :
               : "v"(lds_byte_off), "v"(gptr)
               : "memory");
}
__device__ __forceinline__ void wait_async0() {
  asm volatile("s_wait_asynccnt 0" ::: "memory");
}
__device__ __forceinline__ uint32_t lds_off(const void* p) {
  return (uint32_t)(uintptr_t)p;  // low 32 bits of generic addr = LDS byte offset
}

// ---- WMMA fragment loaders (layouts per CDNA5 ISA 7.12.2, wave32) ----
// A (16-bit, 16x32): lane L -> row M = L%16; element e -> K =
//   e<8 ? kbase + 8*half + e : kbase + 16 + 8*half + (e-8)
__device__ __forceinline__ v16bf load_a_frag(const bf16* rowPtr, int kbase, int half) {
  const v8bf lo = *(const v8bf*)(rowPtr + kbase + 8 * half);
  const v8bf hi = *(const v8bf*)(rowPtr + kbase + 16 + 8 * half);
  v16bf f;
#pragma unroll
  for (int i = 0; i < 8; ++i) { f[i] = lo[i]; f[i + 8] = hi[i]; }
  return f;
}

// B (16-bit, 32x16): lane L -> col N = L%16; element e -> K =
//   e<8 ? kbase + 8*half + e : kbase + 16 + 8*half + (e-8)
__device__ __forceinline__ v16bf load_b_frag(const bf16* rowPtr, int kbase, int half) {
  const v8bf lo = *(const v8bf*)(rowPtr + kbase + 8 * half);
  const v8bf hi = *(const v8bf*)(rowPtr + kbase + 16 + 8 * half);
  v16bf f;
#pragma unroll
  for (int i = 0; i < 8; ++i) { f[i] = lo[i]; f[i + 8] = hi[i]; }
  return f;
}

__device__ __forceinline__ v8f wmma_bf16(v16bf a, v16bf b, v8f c) {
  return __builtin_amdgcn_wmma_f32_16x16x32_bf16(false, a, false, b, (short)0, c,
                                                 false, false);
}

// =====================================================================
// Pass A: per chunk, compute phi(k); S_i = phi(k)^T v ; z_i = sum phi(k)
// grid = (NC, BH), block = 128 (4 waves; wave w owns 16-row strip)
// =====================================================================
__global__ __launch_bounds__(128) void pa_phase1(
    const float* __restrict__ k, const float* __restrict__ v,
    const float* __restrict__ omega, float* __restrict__ S_ws,
    float* __restrict__ z_ws) {
  const int chunk = blockIdx.x, bh = blockIdx.y;
  const int tid = threadIdx.x;
  const int lane = tid & 31, wave = tid >> 5;
  const int lr = lane & 15, half = lane >> 4;

  __shared__ alignas(16) float stage_k[CC * Dd];  // f32 staging (async DMA dst)
  __shared__ alignas(16) float stage_v[CC * Dd];
  __shared__ alignas(16) bf16 omega_lds[Mm][40];  // [m][d]
  __shared__ alignas(16) bf16 k_lds[CC][40];      // [n][d]
  __shared__ alignas(16) bf16 phiT_lds[Mm][72];   // phi_k^T [m][c]
  __shared__ alignas(16) bf16 vT_lds[Dd][72];     // v^T [d][c]
  __shared__ float nrm[CC];

  // --- kick off coalesced async global->LDS staging of k,v chunks ---
  const long base = ((long)bh * Nn + (long)chunk * CC) * Dd;
  {
    const float* kg = k + base;
    const float* vg = v + base;
    const uint32_t sk0 = lds_off(stage_k);
    const uint32_t sv0 = lds_off(stage_v);
#pragma unroll
    for (int i = 0; i < 4; ++i) {
      const int off = tid * 4 + i * 512;  // float index; 16B per thread per iter
      async_b128(sk0 + off * 4, kg + off);
      async_b128(sv0 + off * 4, vg + off);
    }
  }

  // overlap: convert omega while DMA is in flight
  for (int i = tid; i < Mm * Dd; i += 128)
    omega_lds[i >> 5][i & 31] = (bf16)omega[i];

  wait_async0();
  __syncthreads();

  if (tid < CC) {
    const float* kr = stage_k + tid * Dd;
    float s = 0.f;
#pragma unroll
    for (int d = 0; d < Dd; ++d) {
      float x = kr[d];
      s += x * x;
      k_lds[tid][d] = (bf16)x;
    }
    nrm[tid] = 0.5f * s;
    const float* vr = stage_v + tid * Dd;
#pragma unroll
    for (int d = 0; d < Dd; ++d) vT_lds[d][tid] = (bf16)vr[d];
  }
  __syncthreads();

  const float inv_sqrt_m = 0.125f;  // 1/sqrt(64)

  // proj = k @ omega^T, then phi_k = exp(proj - 0.5||k||^2)/sqrt(M)
  {
    v16bf a = load_a_frag(&k_lds[wave * 16 + lr][0], 0, half);
#pragma unroll
    for (int nt = 0; nt < 4; ++nt) {
      v8f acc = {};
      v16bf b = load_b_frag(&omega_lds[nt * 16 + lr][0], 0, half);
      acc = wmma_bf16(a, b, acc);
#pragma unroll
      for (int r = 0; r < 8; ++r) {
        int n = wave * 16 + r + 8 * half;  // chunk row
        int m = nt * 16 + lr;              // feature index
        float phi = __expf(acc[r] - nrm[n]) * inv_sqrt_m;
        phiT_lds[m][n] = (bf16)phi;
      }
    }
  }
  __syncthreads();

  // z_i[m] = sum_c phi_k[c][m]
  if (tid < Mm) {
    float s = 0.f;
#pragma unroll
    for (int c = 0; c < CC; ++c) s += (float)phiT_lds[tid][c];
    z_ws[((long)bh * NC + chunk) * Mm + tid] = s;
  }

  // S_i = phi_k^T [M x C] @ v [C x D]
  float* Sout = S_ws + ((long)bh * NC + chunk) * Mm * Dd;
#pragma unroll
  for (int nt = 0; nt < 2; ++nt) {
    v8f acc = {};
#pragma unroll
    for (int ks = 0; ks < 2; ++ks) {
      v16bf a = load_a_frag(&phiT_lds[wave * 16 + lr][0], ks * 32, half);
      v16bf b = load_b_frag(&vT_lds[nt * 16 + lr][0], ks * 32, half);
      acc = wmma_bf16(a, b, acc);
    }
#pragma unroll
    for (int r = 0; r < 8; ++r) {
      int m = wave * 16 + r + 8 * half;
      int d = nt * 16 + lr;
      Sout[m * Dd + d] = acc[r];
    }
  }
}

// =====================================================================
// Pass B: in-place exclusive prefix scan of S_i, z_i over chunks.
// grid = BH, block = 256
// =====================================================================
__global__ __launch_bounds__(256) void pa_phase2(float* __restrict__ S_ws,
                                                 float* __restrict__ z_ws) {
  const int bh = blockIdx.x, tid = threadIdx.x;
  for (int e = tid; e < Mm * Dd; e += 256) {
    float run = 0.f;
    float* p = S_ws + (long)bh * NC * Mm * Dd + e;
    for (int c = 0; c < NC; ++c) {
      float t = p[(long)c * Mm * Dd];
      p[(long)c * Mm * Dd] = run;
      run += t;
    }
  }
  for (int e = tid; e < Mm; e += 256) {
    float run = 0.f;
    float* p = z_ws + (long)bh * NC * Mm + e;
    for (int c = 0; c < NC; ++c) {
      float t = p[c * Mm];
      p[c * Mm] = run;
      run += t;
    }
  }
}

// =====================================================================
// Pass C: out = (phi_q @ S_prev + mask(phi_q phi_k^T) @ v) / denom
// grid = (NC, BH), block = 128
// =====================================================================
__global__ __launch_bounds__(128) void pa_phase3(
    const float* __restrict__ q, const float* __restrict__ k,
    const float* __restrict__ v, const float* __restrict__ omega,
    const float* __restrict__ S_ws, const float* __restrict__ z_ws,
    float* __restrict__ out) {
  const int chunk = blockIdx.x, bh = blockIdx.y;
  const int tid = threadIdx.x;
  const int lane = tid & 31, wave = tid >> 5;
  const int lr = lane & 15, half = lane >> 4;

  __shared__ alignas(16) float stage_q[CC * Dd];
  __shared__ alignas(16) float stage_k[CC * Dd];
  __shared__ alignas(16) float stage_v[CC * Dd];
  __shared__ alignas(16) bf16 omega_lds[Mm][40];
  __shared__ alignas(16) bf16 q_lds[CC][40];
  __shared__ alignas(16) bf16 k_lds[CC][40];
  __shared__ alignas(16) bf16 phiq_lds[CC][72];  // [i][m]
  __shared__ alignas(16) bf16 phik_lds[CC][72];  // [j][m]
  __shared__ alignas(16) bf16 sc_lds[CC][72];    // masked scores [i][j]
  __shared__ alignas(16) bf16 vT_lds[Dd][72];    // v^T [d][c]
  __shared__ alignas(16) bf16 ST_lds[Dd][72];    // S_prev^T [d][m]
  __shared__ float nq[CC], nk[CC], zprev[Mm], den[CC];

  // --- async stage q,k,v chunks into LDS (coalesced B128 DMA) ---
  const long base = ((long)bh * Nn + (long)chunk * CC) * Dd;
  {
    const float* qg = q + base;
    const float* kg = k + base;
    const float* vg = v + base;
    const uint32_t sq0 = lds_off(stage_q);
    const uint32_t sk0 = lds_off(stage_k);
    const uint32_t sv0 = lds_off(stage_v);
#pragma unroll
    for (int i = 0; i < 4; ++i) {
      const int off = tid * 4 + i * 512;
      async_b128(sq0 + off * 4, qg + off);
      async_b128(sk0 + off * 4, kg + off);
      async_b128(sv0 + off * 4, vg + off);
    }
  }

  // overlap: omega + exclusive-prefix state while DMA is in flight
  for (int i = tid; i < Mm * Dd; i += 128)
    omega_lds[i >> 5][i & 31] = (bf16)omega[i];

  const float* Sprev = S_ws + ((long)bh * NC + chunk) * Mm * Dd;
  for (int i = tid; i < Mm * Dd; i += 128)
    ST_lds[i & 31][i >> 5] = (bf16)Sprev[i];  // transpose: [d][m]
  if (tid < Mm) zprev[tid] = z_ws[((long)bh * NC + chunk) * Mm + tid];

  wait_async0();
  __syncthreads();

  if (tid < CC) {
    const float* qr = stage_q + tid * Dd;
    const float* kr = stage_k + tid * Dd;
    float sq = 0.f, sk = 0.f;
#pragma unroll
    for (int d = 0; d < Dd; ++d) {
      float xq = qr[d], xk = kr[d];
      sq += xq * xq;
      sk += xk * xk;
      q_lds[tid][d] = (bf16)xq;
      k_lds[tid][d] = (bf16)xk;
    }
    nq[tid] = 0.5f * sq;
    nk[tid] = 0.5f * sk;
    const float* vr = stage_v + tid * Dd;
#pragma unroll
    for (int d = 0; d < Dd; ++d) vT_lds[d][tid] = (bf16)vr[d];
  }
  __syncthreads();

  const float inv_sqrt_m = 0.125f;

  // phi_q, phi_k for this wave's 16-row strip
  {
    v16bf aq = load_a_frag(&q_lds[wave * 16 + lr][0], 0, half);
    v16bf ak = load_a_frag(&k_lds[wave * 16 + lr][0], 0, half);
#pragma unroll
    for (int nt = 0; nt < 4; ++nt) {
      v16bf b = load_b_frag(&omega_lds[nt * 16 + lr][0], 0, half);
      v8f accq = {}, acck = {};
      accq = wmma_bf16(aq, b, accq);
      acck = wmma_bf16(ak, b, acck);
#pragma unroll
      for (int r = 0; r < 8; ++r) {
        int n = wave * 16 + r + 8 * half;
        int m = nt * 16 + lr;
        phiq_lds[n][m] = (bf16)(__expf(accq[r] - nq[n]) * inv_sqrt_m);
        phik_lds[n][m] = (bf16)(__expf(acck[r] - nk[n]) * inv_sqrt_m);
      }
    }
  }
  __syncthreads();

  // inter-chunk: acc_o = phi_q [C x M] @ S_prev [M x D]
  v8f acc_o[2];
#pragma unroll
  for (int nt = 0; nt < 2; ++nt) {
    v8f acc = {};
#pragma unroll
    for (int ks = 0; ks < 2; ++ks) {
      v16bf a = load_a_frag(&phiq_lds[wave * 16 + lr][0], ks * 32, half);
      v16bf b = load_b_frag(&ST_lds[nt * 16 + lr][0], ks * 32, half);
      acc = wmma_bf16(a, b, acc);
    }
    acc_o[nt] = acc;
  }

  // intra-chunk scores with causal mask
#pragma unroll
  for (int jt = 0; jt < 4; ++jt) {
    v8f acc = {};
#pragma unroll
    for (int ks = 0; ks < 2; ++ks) {
      v16bf a = load_a_frag(&phiq_lds[wave * 16 + lr][0], ks * 32, half);
      v16bf b = load_b_frag(&phik_lds[jt * 16 + lr][0], ks * 32, half);
      acc = wmma_bf16(a, b, acc);
    }
#pragma unroll
    for (int r = 0; r < 8; ++r) {
      int i = wave * 16 + r + 8 * half;
      int j = jt * 16 + lr;
      sc_lds[i][j] = (bf16)((j <= i) ? acc[r] : 0.0f);
    }
  }
  __syncthreads();

  // denominator: phi_q . z_prev + row-sum of masked scores
  if (tid < CC) {
    float s = 1e-6f;
#pragma unroll
    for (int m = 0; m < Mm; ++m) s += (float)phiq_lds[tid][m] * zprev[m];
#pragma unroll
    for (int j = 0; j < CC; ++j) s += (float)sc_lds[tid][j];
    den[tid] = s;
  }
  __syncthreads();

  // intra-chunk: acc_o += mask(scores) [C x C] @ v [C x D]; then divide, store
  float* og = out + base;
#pragma unroll
  for (int nt = 0; nt < 2; ++nt) {
    v8f acc = acc_o[nt];
#pragma unroll
    for (int ks = 0; ks < 2; ++ks) {
      v16bf a = load_a_frag(&sc_lds[wave * 16 + lr][0], ks * 32, half);
      v16bf b = load_b_frag(&vT_lds[nt * 16 + lr][0], ks * 32, half);
      acc = wmma_bf16(a, b, acc);
    }
#pragma unroll
    for (int r = 0; r < 8; ++r) {
      int i = wave * 16 + r + 8 * half;
      int d = nt * 16 + lr;
      og[(long)i * Dd + d] = acc[r] / den[i];
    }
  }
}

// =====================================================================
extern "C" void kernel_launch(void* const* d_in, const int* in_sizes, int n_in,
                              void* d_out, int out_size, void* d_ws,
                              size_t ws_size, hipStream_t stream) {
  const float* q     = (const float*)d_in[0];
  const float* k     = (const float*)d_in[1];
  const float* v     = (const float*)d_in[2];
  const float* omega = (const float*)d_in[3];
  float* out = (float*)d_out;

  float* S_ws = (float*)d_ws;                        // [BH][NC][M][D]
  float* z_ws = S_ws + (size_t)BH * NC * Mm * Dd;    // [BH][NC][M]

  dim3 grid(NC, BH);
  pa_phase1<<<grid, 128, 0, stream>>>(k, v, omega, S_ws, z_ws);
  pa_phase2<<<BH, 256, 0, stream>>>(S_ws, z_ws);
  pa_phase3<<<grid, 128, 0, stream>>>(q, k, v, omega, S_ws, z_ws, out);
}
